// KAN_linear_21423296873081
// MI455X (gfx1250) — matrix operand order repaired and checked by reading, not compiled, not run on hardware
//
#include <hip/hip_runtime.h>

typedef __attribute__((ext_vector_type(16))) _Float16 v16h;
typedef __attribute__((ext_vector_type(8)))  float    v8f;

#define INDIM   128
#define OUTDIM  256
#define GRID_SZ 16
#define KTOT    4096      // 2 (cos/sin) * 128 (indim) * 16 (grid)
#define NSTEP   (KTOT / 32)
#define M_TILE  128
#define XPAD    132       // LDS row stride (floats): 16B-aligned, 4-bank stride
#define INV2PI  0.15915494309189535f

// ---------------------------------------------------------------------------
// Pre-pass: fouriercoeffs f32 (2, 256, 128, 16) -> f16 B matrix Bh[o][k],
// k = t*2048 + i*16 + g, so a lane's WMMA B fragment is two 16-byte chunks, 32 bytes apart.
// ---------------------------------------------------------------------------
__global__ __launch_bounds__(256) void convert_coeffs(
    const float* __restrict__ c, _Float16* __restrict__ bh) {
  int idx = blockIdx.x * 256 + threadIdx.x;   // idx = o*4096 + k
  int o = idx >> 12;
  int k = idx & (KTOT - 1);
  int t = k >> 11;
  int r = k & 2047;
  int i = r >> 4;
  int g = r & 15;
  bh[idx] = (_Float16)c[(((t * OUTDIM) + o) * INDIM + i) * GRID_SZ + g];
}

// ---- helpers --------------------------------------------------------------
__device__ __forceinline__ void loadB8(v16h (&buf)[8], const _Float16* base) {
#pragma unroll
  for (int nf = 0; nf < 8; ++nf)
  {
    typedef __attribute__((ext_vector_type(8))) _Float16 v8h;
    const _Float16* p = base + (size_t)nf * 16 * KTOT;
    v8h lo = *(const v8h*)p;
    v8h hi = *(const v8h*)(p + 16);
    buf[nf] = __builtin_shufflevector(lo, hi, 0, 1, 2, 3, 4, 5, 6, 7, 8, 9, 10, 11, 12, 13, 14, 15);
  }
}

template <int IS_SIN>
__device__ __forceinline__ v16h trigA(float x0n, float x1n, int kf0) {
  // A fragment, 16-bit 16x32 layout: element e -> K = (e<8?0:16) + 8h + (e&7)
  // x already normalized by 1/2pi; v_sin/v_cos take revolutions.
  v16h a;
#pragma unroll
  for (int e = 0; e < 16; ++e) {
    float xv  = (e & 8) ? x1n : x0n;
    float ang = xv * (float)(kf0 + (e & 7));
    float v   = IS_SIN ? __builtin_amdgcn_sinf(ang) : __builtin_amdgcn_cosf(ang);
    a[e] = (_Float16)v;
  }
  return a;
}

template <int OFF>
__device__ __forceinline__ void wmma8(v8f (&acc)[16], v16h a, const v16h (&b)[8]) {
#pragma unroll
  for (int nf = 0; nf < 8; ++nf)
    acc[OFF + nf] = __builtin_amdgcn_wmma_f32_16x16x32_f16(
        false, a, false, b[nf], (short)0, acc[OFF + nf], false, false);
}

// One trig phase (cos or sin), 64 K-steps, software-pipelined at 8-fragment
// granularity.  bA holds frags 0..7 of the *current* step on entry.
template <int IS_SIN>
__device__ __forceinline__ void phase(v8f (&acc)[16], v16h (&bA)[8], v16h (&bB)[8],
                                      const _Float16* bcol, const float* xrow,
                                      int kf0, int s0) {
#pragma unroll 1
  for (int s = s0; s < s0 + 64; ++s) {
    const int i0 = (s & 63) * 2;
    const _Float16* bstep = bcol + (size_t)(s & (NSTEP - 1)) * 32;
    const _Float16* bnext = bcol + (size_t)((s + 1) & (NSTEP - 1)) * 32;

    loadB8(bB, bstep + (size_t)8 * 16 * KTOT);   // frags 8..15 of step s
    __builtin_amdgcn_sched_barrier(0);           // pin loads above the chain

    v16h a = trigA<IS_SIN>(xrow[i0], xrow[i0 + 1], kf0);
    wmma8<0>(acc, a, bA);                        // consume bA (loaded last iter)
    loadB8(bA, bnext);                           // frags 0..7 of step s+1
    __builtin_amdgcn_sched_barrier(0);

    wmma8<8>(acc, a, bB);                        // consume bB under bA's latency
  }
}

// ---------------------------------------------------------------------------
// Fused KAN-Fourier GEMM: y = trig(x ⊗ k) @ coeffs + bias
// 256 threads = 8 wave32 waves; each wave: 16 rows x 256 cols (16 WMMA accs).
// ---------------------------------------------------------------------------
__global__ __launch_bounds__(256) void kan_fourier_wmma(
    const float* __restrict__ x, const _Float16* __restrict__ bh,
    const float* __restrict__ bias, float* __restrict__ out) {
  __shared__ __align__(16) float xs[M_TILE][XPAD];

  const int tid     = threadIdx.x;
  const int lane    = tid & 31;
  const int wid     = tid >> 5;
  const int rowBase = blockIdx.x * M_TILE;

  // stage x tile (128 x 128 f32) into LDS, pre-normalized by 1/2pi
#pragma unroll
  for (int it = 0; it < 16; ++it) {
    int idx = it * 256 + tid;
    int rr  = idx >> 5;
    int c4  = idx & 31;
    float4 v = ((const float4*)(x + (size_t)(rowBase + rr) * INDIM))[c4];
    v.x *= INV2PI; v.y *= INV2PI; v.z *= INV2PI; v.w *= INV2PI;
    *(float4*)&xs[rr][c4 * 4] = v;
  }
  __syncthreads();

  const int m    = lane & 15;
  const int h    = lane >> 4;
  const int mRow = wid * 16 + m;
  const int kf0  = 8 * h + 1;

  v8f acc[16];
  const v8f vzero = {0.f, 0.f, 0.f, 0.f, 0.f, 0.f, 0.f, 0.f};
#pragma unroll
  for (int nf = 0; nf < 16; ++nf) acc[nf] = vzero;

  const _Float16* bcol = bh + (size_t)m * KTOT + h * 8;
  const float*    xrow = &xs[mRow][0];

  v16h bA[8], bB[8];
  loadB8(bA, bcol);                              // frags 0..7 of step 0

  phase<0>(acc, bA, bB, bcol, xrow, kf0, 0);     // cos: steps 0..63
  phase<1>(acc, bA, bB, bcol, xrow, kf0, 64);    // sin: steps 64..127

  // epilogue: D layout lane L: col = L%16, VGPR e -> M = e + 8*(L/16)
#pragma unroll
  for (int nf = 0; nf < 16; ++nf) {
    int   col = nf * 16 + m;
    float bv  = bias[col];
#pragma unroll
    for (int e = 0; e < 8; ++e) {
      int row = rowBase + wid * 16 + e + 8 * h;
      out[(size_t)row * OUTDIM + col] = acc[nf][e] + bv;
    }
  }
}

// ---------------------------------------------------------------------------
extern "C" void kernel_launch(void* const* d_in, const int* in_sizes, int n_in,
                              void* d_out, int out_size, void* d_ws, size_t ws_size,
                              hipStream_t stream) {
  const float* x      = (const float*)d_in[0];   // (32768, 128) f32
  const float* coeffs = (const float*)d_in[1];   // (2, 256, 128, 16) f32
  const float* bias   = (const float*)d_in[2];   // (1, 256) f32
  float*       out    = (float*)d_out;           // (32768, 256) f32
  _Float16*    bh     = (_Float16*)d_ws;         // 256*4096 f16 = 2 MB scratch

  int nrows = in_sizes[0] / INDIM;               // 32768

  convert_coeffs<<<(OUTDIM * KTOT) / 256, 256, 0, stream>>>(coeffs, bh);
  kan_fourier_wmma<<<nrows / M_TILE, 256, 0, stream>>>(x, bh, bias, out);
}
